// OuterProductDecoder_56066503082146
// MI455X (gfx1250) — compile-verified, matrix operand order checked by reading, not run on hardware
//
#include <hip/hip_runtime.h>

typedef __attribute__((ext_vector_type(16))) __bf16 v16bf;
typedef __attribute__((ext_vector_type(8)))  float  v8f;
typedef unsigned int u32x4 __attribute__((ext_vector_type(4)));
typedef int          i32x8 __attribute__((ext_vector_type(8)));
typedef int          i32x4 __attribute__((ext_vector_type(4)));

#define B_  128
#define D_  256
#define R_  128
#define T_  128

// ---- workspace layout (float elements) ----
#define WS_Q     0          // 32768  query [128][256]
#define WS_K     32768      // 32768  key   [128][256]
#define WS_XG    65536      // 98304  x_gates [128][768]  (r|z|n)
#define WS_CE    163840     // 32768  context_embed [128][256]
#define WS_H0    196608     // 32768  hidden buf 0
#define WS_H1    229376     // 32768  hidden buf 1
#define WS_LOG   262144     // 16384  logits [128][128]
#define WS_HG    278528     // 98304  h_gates [128][768]
#define WS_USED  376832     // 512 u32 (used bitmask, 4 words/row)
#define WS_SYNC  377344     // cnt, gen
#define WS_PART  377856     // 16*32*128*16 = 1048576 split-K partials

// ---- output layout (elements, concatenated return tuple) ----
#define O_ACT  0            // actions [128][128] (int32 bits)
#define O_MASK 16384        // mask_scores [128][128][128]
#define O_S    2113536      // s_list [128][128][256]
#define O_H    6307840      // h_list
#define O_C    10502144     // c_list

// K-offset of element e (of the v16bf operand) for 16-bit WMMA A/B layout:
// lanes 0-15: V0..V3 -> K 0..7, V4..V7 -> K 16..23; lanes 16-31: +8
__device__ __forceinline__ int wmma_koff(int e, int hi) {
  int r = e >> 1;
  return hi * 8 + (r >> 2) * 16 + (r & 3) * 2 + (e & 1);
}

// elements along contiguous k within one row: rowp[k0 + koff]
__device__ __forceinline__ v16bf ld_bf16_kcontig(const float* rowp, int k0, int hi) {
  v16bf v;
#pragma unroll
  for (int e = 0; e < 16; ++e) v[e] = (__bf16)rowp[k0 + wmma_koff(e, hi)];
  return v;
}

// elements strided along k: P[(k0+koff)*stride + col]  (row-major KxN B-matrix)
__device__ __forceinline__ v16bf ld_bf16_kstrided(const float* P, int k0, int stride,
                                                  int col, int hi) {
  v16bf v;
#pragma unroll
  for (int e = 0; e < 16; ++e) v[e] = (__bf16)P[(size_t)(k0 + wmma_koff(e, hi)) * stride + col];
  return v;
}

__device__ __forceinline__ v8f wmma_bf16(v16bf a, v16bf b, v8f c) {
  return __builtin_amdgcn_wmma_f32_16x16x32_bf16(false, a, false, b, (short)0, c, false, false);
}

// ---- Tensor Data Mover: 1-D fp32 copy global -> LDS (clang-23 6-arg builtin) ----
// D# per CDNA5 ISA: group0 = {flags, lds_addr, global_addr, type=2},
// group1 = {data_size=4B, tensor_dim0=n, tensor_dim1=1, tile_dim0=n, tile_dim1=1, stride0=n}.
// Groups 2/3 (and the extra group) zero: tensor is <= 2-D.
__device__ __forceinline__ void tdm_load_f32_1d(unsigned lds_byte_off, const float* gptr,
                                                unsigned n /* fp32 elements, < 65536 */) {
  unsigned long long ga = (unsigned long long)(uintptr_t)gptr;
  u32x4 g0;
  g0[0] = 1u;                                         // count=1 (valid), no gather/restore
  g0[1] = lds_byte_off;                               // lds_addr (bytes)
  g0[2] = (unsigned)(ga & 0xFFFFFFFFu);               // global_addr[31:0]
  g0[3] = (unsigned)((ga >> 32) & 0x1FFFFFFu) | (2u << 30);  // global_addr[56:32] | type=2
  i32x8 g1;
  g1[0] = (int)(2u << 16);                            // workgroup_mask=0, data_size=2 (4B)
  g1[1] = (int)((n & 0xFFFFu) << 16);                 // tensor_dim0[15:0]
  g1[2] = (int)(((n >> 16) & 0xFFFFu) | (1u << 16));  // tensor_dim0[31:16], tensor_dim1=1
  g1[3] = (int)((n & 0xFFFFu) << 16);                 // tile_dim0 = n
  g1[4] = 1;                                          // tile_dim1=1, tile_dim2=0
  g1[5] = (int)n;                                     // tensor_dim0_stride[31:0]
  g1[6] = 0;                                          // stride0 hi, stride1 lo
  g1[7] = 0;
  i32x4 gz4 = {0, 0, 0, 0};
  i32x8 gz8 = {0, 0, 0, 0, 0, 0, 0, 0};
  __builtin_amdgcn_tensor_load_to_lds(g0, g1, gz4, gz4, gz8, 0 /* cpol */);
}

// ---------------- K1: query = h@Wq+bq ; key = h@Wk+bk ----------------
__global__ __launch_bounds__(256) void k_qk(const float* __restrict__ h,
                                            const float* __restrict__ Wq, const float* __restrict__ bq,
                                            const float* __restrict__ Wk, const float* __restrict__ bk,
                                            float* __restrict__ ws) {
  int w = blockIdx.x * 8 + (threadIdx.x >> 5);      // 0..255 : 128 Q tiles, 128 K tiles
  int lane = threadIdx.x & 31, ln = lane & 15, hi = lane >> 4;
  const float* Wm   = (w < 128) ? Wq : Wk;
  const float* bias = (w < 128) ? bq : bk;
  float* out = ws + ((w < 128) ? WS_Q : WS_K);
  int tw = w & 127;
  int mbase = (tw >> 4) * 16, nbase = (tw & 15) * 16;
  v8f c = {};
  const float* arow = h + (size_t)(mbase + ln) * D_;
#pragma unroll
  for (int k0 = 0; k0 < D_; k0 += 32) {
    v16bf a = ld_bf16_kcontig(arow, k0, hi);
    v16bf b = ld_bf16_kstrided(Wm, k0, D_, nbase + ln, hi);
    c = wmma_bf16(a, b, c);
  }
  float bv = bias[nbase + ln];
#pragma unroll
  for (int r = 0; r < 8; ++r)
    out[(size_t)(mbase + r + 8 * hi) * D_ + nbase + ln] = c[r] + bv;
}

// ---------------- K2: x_gates = key @ W_ih^T + b_ih  [128][768] ----------------
__global__ __launch_bounds__(256) void k_xg(const float* __restrict__ W_ih,
                                            const float* __restrict__ b_ih,
                                            float* __restrict__ ws) {
  int w = blockIdx.x * 8 + (threadIdx.x >> 5);      // 0..383 : 8 x 48 tiles
  int lane = threadIdx.x & 31, ln = lane & 15, hi = lane >> 4;
  int mbase = (w / 48) * 16, nbase = (w % 48) * 16;
  const float* key = ws + WS_K;
  float* out = ws + WS_XG;
  v8f c = {};
  const float* arow = key + (size_t)(mbase + ln) * D_;
  const float* brow = W_ih + (size_t)(nbase + ln) * D_;   // B[k,n] = W_ih[n*D+k]
#pragma unroll
  for (int k0 = 0; k0 < D_; k0 += 32) {
    v16bf a = ld_bf16_kcontig(arow, k0, hi);
    v16bf b = ld_bf16_kcontig(brow, k0, hi);
    c = wmma_bf16(a, b, c);
  }
  float bv = b_ih[nbase + ln];
#pragma unroll
  for (int r = 0; r < 8; ++r)
    out[(size_t)(mbase + r + 8 * hi) * 768 + nbase + ln] = c[r] + bv;
}

// ---------------- K3: big GEMM, split-K: streams Wc (64MB) once ----------------
// context[b, i*256+j] = q[b,i]*key[b,j] -> A tiles synthesized from q / TDM-staged key.
// Grid: 16 oj-tiles x 16 ksegs x 2 batch-halves = 512 blocks.
// Per block: TDM DMA of a 64KB fp32 key slice into LDS; 8 waves = 4 b-tiles x 2 k-subsegs.
__global__ __launch_bounds__(256) void k_ctx(const float* __restrict__ Wc,
                                             float* __restrict__ ws) {
  __shared__ float skey[64 * D_];                   // 64 KB fp32 key slice
  const float* key = ws + WS_K;
  const float* q   = ws + WS_Q;

  int oj    = blockIdx.x >> 5;                      // 16 output-column tiles
  int rem   = blockIdx.x & 31;
  int kseg16 = rem >> 1;                            // 16 coarse K segments
  int bhalf  = rem & 1;                             // batch rows [bhalf*64, bhalf*64+64)

  if ((threadIdx.x >> 5) == 0) {                    // one wave issues the DMA
    tdm_load_f32_1d((unsigned)(uintptr_t)skey,
                    key + (size_t)bhalf * 64 * D_, 64 * D_);
    __builtin_amdgcn_s_wait_tensorcnt(0);
  }
  __syncthreads();

  int wave = threadIdx.x >> 5, lane = threadIdx.x & 31;
  int ln = lane & 15, hi = lane >> 4;
  int lt   = wave >> 1;                             // local b-tile 0..3
  int subk = wave & 1;                              // k subsegment within kseg16
  int seg32 = kseg16 * 2 + subk;                    // effective 32-way K split (as before)
  int btile = lt + bhalf * 4;
  int mbase = btile * 16, obase = oj * 16;
  int m  = mbase + ln;                              // global batch row
  int lm = lt * 16 + ln;                            // row within LDS slice
  v8f c = {};
  for (int ch = 0; ch < 64; ++ch) {                 // 64 chunks of K=32 per segment
    int k0 = seg32 * 2048 + ch * 32;
    int i = k0 >> 8, jb = k0 & 255;                 // chunk lies within one i-row
    if (ch < 60)                                    // keep the 64MB Wc stream deep in flight
      __builtin_prefetch(&Wc[(size_t)(k0 + 128) * D_ + obase + ln], 0, 1);
    float qf = q[(size_t)m * D_ + i];
    v16bf a, b;
#pragma unroll
    for (int e = 0; e < 16; ++e) {
      int ko = wmma_koff(e, hi);
      a[e] = (__bf16)(qf * skey[lm * D_ + jb + ko]);
      b[e] = (__bf16)Wc[(size_t)(k0 + ko) * D_ + obase + ln];
    }
    c = wmma_bf16(a, b, c);
  }
  float* part = ws + WS_PART;
#pragma unroll
  for (int r = 0; r < 8; ++r)
    part[(size_t)((oj * 32 + seg32) * 128 + (mbase + r + 8 * hi)) * 16 + ln] = c[r];
}

// ---------------- K4: deterministic split-K reduce + state init ----------------
__global__ __launch_bounds__(256) void k_reduce_init(const float* __restrict__ bc,
                                                     const float* __restrict__ h,
                                                     float* __restrict__ ws) {
  int tid = blockIdx.x * 256 + threadIdx.x;         // 0..32767
  int b = tid >> 8, o = tid & 255;
  int ojn = o >> 4, n = o & 15;
  const float* part = ws + WS_PART;
  float s = 0.f;
#pragma unroll 4
  for (int seg = 0; seg < 32; ++seg)
    s += part[(size_t)((ojn * 32 + seg) * 128 + b) * 16 + n];
  ws[WS_CE + tid] = s + bc[o];
  ws[WS_H0 + tid] = h[tid];                         // hidden_0 = h
  if (tid < 512) ((unsigned*)(ws + WS_USED))[tid] = 0u;
  if (tid < 2)   ((unsigned*)(ws + WS_SYNC))[tid] = 0u;
}

// ---------------- K5: s_list broadcast (loop-invariant context_embed) ----------------
__global__ __launch_bounds__(256) void k_slist(float* __restrict__ out,
                                               const float* __restrict__ ws) {
  size_t idx = (size_t)blockIdx.x * 256 + threadIdx.x;   // < 4194304
  int d = (int)(idx & 255);
  int b = (int)(idx >> 15);                              // / (T*D)
  out[O_S + idx] = ws[WS_CE + b * 256 + d];
}

// ---------------- software grid barrier (persistent kernel) ----------------
__device__ __forceinline__ void grid_barrier(unsigned* cnt, unsigned* gen, unsigned nb) {
  __threadfence();
  __syncthreads();
  if (threadIdx.x == 0) {
    unsigned g = __hip_atomic_load(gen, __ATOMIC_RELAXED, __HIP_MEMORY_SCOPE_AGENT);
    unsigned arrived = __hip_atomic_fetch_add(cnt, 1u, __ATOMIC_ACQ_REL, __HIP_MEMORY_SCOPE_AGENT);
    if (arrived == nb - 1) {
      __hip_atomic_store(cnt, 0u, __ATOMIC_RELAXED, __HIP_MEMORY_SCOPE_AGENT);
      __hip_atomic_fetch_add(gen, 1u, __ATOMIC_RELEASE, __HIP_MEMORY_SCOPE_AGENT);
    } else {
      while (__hip_atomic_load(gen, __ATOMIC_ACQUIRE, __HIP_MEMORY_SCOPE_AGENT) == g)
        __builtin_amdgcn_s_sleep(1);
    }
  }
  __syncthreads();
}

// ---------------- K6: persistent sequential scan, T=128 steps ----------------
// 64 blocks x 8 waves = 512 waves. Phase1: 448 WMMA tiles (logits + h_gates) +
// trajectory writes. Phase2: per-row masked argmax + GRU update (double-buffered).
__global__ __launch_bounds__(256) void k_scan(const float* __restrict__ Wa,
                                              const float* __restrict__ ba,
                                              const float* __restrict__ W_hh,
                                              const float* __restrict__ b_hh,
                                              float* __restrict__ ws,
                                              float* __restrict__ out) {
  unsigned* cnt  = (unsigned*)(ws + WS_SYNC);
  unsigned* gen  = cnt + 1;
  unsigned* used = (unsigned*)(ws + WS_USED);
  int w = blockIdx.x * 8 + (threadIdx.x >> 5);      // 0..511
  int lane = threadIdx.x & 31, ln = lane & 15, hi = lane >> 4;
  int tg = blockIdx.x * 256 + threadIdx.x;          // 0..16383
  float* logits = ws + WS_LOG;
  float* hg     = ws + WS_HG;
  const float* xg = ws + WS_XG;
  int* actions = (int*)out;

  for (int t = 0; t < T_; ++t) {
    const float* hid = ws + ((t & 1) ? WS_H1 : WS_H0);
    float* hnext     = ws + ((t & 1) ? WS_H0 : WS_H1);

    // ---- phase 1: both GEMMs read the same hidden ----
    if (w < 64) {                                   // logits = hid @ Wa + ba   [128][128]
      int mbase = (w >> 3) * 16, nbase = (w & 7) * 16;
      v8f c = {};
      const float* arow = hid + (size_t)(mbase + ln) * D_;
#pragma unroll
      for (int k0 = 0; k0 < D_; k0 += 32) {
        v16bf a = ld_bf16_kcontig(arow, k0, hi);
        v16bf b = ld_bf16_kstrided(Wa, k0, R_, nbase + ln, hi);
        c = wmma_bf16(a, b, c);
      }
      float bv = ba[nbase + ln];
#pragma unroll
      for (int r = 0; r < 8; ++r)
        logits[(mbase + r + 8 * hi) * R_ + nbase + ln] = c[r] + bv;
    } else if (w < 448) {                           // h_gates = hid @ W_hh^T + b_hh [128][768]
      int wh = w - 64;
      int mbase = (wh / 48) * 16, nbase = (wh % 48) * 16;
      v8f c = {};
      const float* arow = hid + (size_t)(mbase + ln) * D_;
      const float* brow = W_hh + (size_t)(nbase + ln) * D_;
#pragma unroll
      for (int k0 = 0; k0 < D_; k0 += 32) {
        v16bf a = ld_bf16_kcontig(arow, k0, hi);
        v16bf b = ld_bf16_kcontig(brow, k0, hi);
        c = wmma_bf16(a, b, c);
      }
      float bv = b_hh[nbase + ln];
#pragma unroll
      for (int r = 0; r < 8; ++r)
        hg[(mbase + r + 8 * hi) * 768 + nbase + ln] = c[r] + bv;
    }
    // trajectory: h_list/c_list store the pre-update hidden
#pragma unroll
    for (int u = 0; u < 2; ++u) {
      int e = tg * 2 + u;
      int b = e >> 8, d = e & 255;
      float hv = hid[e];
      size_t o = (size_t)(b * T_ + t) * D_ + d;
      out[O_H + o] = hv;
      out[O_C + o] = hv;
    }
    grid_barrier(cnt, gen, gridDim.x);

    // ---- phase 2a: masked argmax per batch row (wave32 reduction) ----
    if (w < 128) {
      int b = w;
      float bv = -__builtin_inff();
      int bi = 0x7fffffff;
#pragma unroll
      for (int u = 0; u < 4; ++u) {
        int col = lane + u * 32;
        unsigned ub = used[b * 4 + (col >> 5)];
        float lv = logits[b * R_ + col];
        float mv = ((ub >> (col & 31)) & 1u) ? -__builtin_inff() : lv;
        out[O_MASK + (size_t)(b * T_ + t) * R_ + col] = mv;
        if (mv > bv || (mv == bv && col < bi)) { bv = mv; bi = col; }
      }
#pragma unroll
      for (int off = 16; off > 0; off >>= 1) {       // first-max wins (matches jnp.argmax)
        float ov = __shfl_xor(bv, off, 32);
        int   oi = __shfl_xor(bi, off, 32);
        if (ov > bv || (ov == bv && oi < bi)) { bv = ov; bi = oi; }
      }
      if (lane == 0) {
        actions[b * T_ + t] = bi;
        used[b * 4 + (bi >> 5)] |= 1u << (bi & 31);
      }
    }
    // ---- phase 2b: GRU elementwise update ----
#pragma unroll
    for (int u = 0; u < 2; ++u) {
      int e = tg * 2 + u;
      int b = e >> 8, d = e & 255;
      float xr = xg[b * 768 + d], xz = xg[b * 768 + 256 + d], xn = xg[b * 768 + 512 + d];
      float hr = hg[b * 768 + d], hz = hg[b * 768 + 256 + d], hn = hg[b * 768 + 512 + d];
      float rr = 1.f / (1.f + __expf(-(xr + hr)));
      float zz = 1.f / (1.f + __expf(-(xz + hz)));
      float nn = tanhf(xn + rr * hn);
      hnext[e] = (1.f - zz) * nn + zz * hid[e];
    }
    grid_barrier(cnt, gen, gridDim.x);
  }
}

extern "C" void kernel_launch(void* const* d_in, const int* in_sizes, int n_in,
                              void* d_out, int out_size, void* d_ws, size_t ws_size,
                              hipStream_t stream) {
  (void)in_sizes; (void)n_in; (void)out_size; (void)ws_size;
  const float* h    = (const float*)d_in[0];
  const float* Wq   = (const float*)d_in[1];
  const float* bq   = (const float*)d_in[2];
  const float* Wk   = (const float*)d_in[3];
  const float* bk   = (const float*)d_in[4];
  const float* Wc   = (const float*)d_in[5];
  const float* bc   = (const float*)d_in[6];
  const float* Wa   = (const float*)d_in[7];
  const float* ba   = (const float*)d_in[8];
  const float* W_ih = (const float*)d_in[9];
  const float* W_hh = (const float*)d_in[10];
  const float* b_ih = (const float*)d_in[11];
  const float* b_hh = (const float*)d_in[12];
  float* ws  = (float*)d_ws;
  float* out = (float*)d_out;

  k_qk<<<32, 256, 0, stream>>>(h, Wq, bq, Wk, bk, ws);
  k_xg<<<48, 256, 0, stream>>>(W_ih, b_ih, ws);
  k_ctx<<<512, 256, 0, stream>>>(Wc, ws);
  k_reduce_init<<<128, 256, 0, stream>>>(bc, h, ws);
  k_slist<<<16384, 256, 0, stream>>>(out, ws);
  k_scan<<<64, 256, 0, stream>>>(Wa, ba, W_hh, b_hh, ws, out);
}
